// VisionMamba3D_11708080849429
// MI455X (gfx1250) — matrix operand order repaired and checked by reading, not compiled, not run on hardware
//
#include <hip/hip_runtime.h>
#include <hip/hip_bf16.h>
#include <math.h>

// ---------------------------------------------------------------------------
// Types for CDNA5 WMMA
// ---------------------------------------------------------------------------
typedef __attribute__((ext_vector_type(16))) __bf16 v16bf;
typedef __attribute__((ext_vector_type(8)))  float  v8f;

#define ACT_NONE     0
#define ACT_RELU     1
#define ACT_GELU     2
#define ACT_SILU     3
#define ACT_SOFTPLUS 4

__device__ __forceinline__ unsigned short f2bf(float f) {
  unsigned int u = __float_as_uint(f);
  u += 0x7FFFu + ((u >> 16) & 1u);          // round-to-nearest-even
  return (unsigned short)(u >> 16);
}

__device__ __forceinline__ unsigned int pack2bf(float lo, float hi) {
  return (unsigned int)f2bf(lo) | ((unsigned int)f2bf(hi) << 16);
}

__device__ __forceinline__ float act_apply(float x, int act) {
  switch (act) {
    case ACT_RELU:     return fmaxf(x, 0.0f);
    case ACT_GELU:     return 0.5f * x * (1.0f + erff(x * 0.7071067811865476f));
    case ACT_SILU:     return x / (1.0f + __expf(-x));
    case ACT_SOFTPLUS: return (x > 20.0f) ? x : log1pf(__expf(x));
    default:           return x;
  }
}

// ---------------------------------------------------------------------------
// Generic GEMM:  C[M,N] = act( A[M,K] @ W[N,K]^T + bias[N] )
// Tile: 128(M) x 64(N), K staged 32 wide through LDS as bf16.
// 256 threads = 8 wave32 waves; each wave owns a 32x32 output
// (2 A-frags x 2 B-frags = 4 v_wmma_f32_16x16x32_bf16 per K-step).
// Fast staging path: uniform (scalar-branch) condition, float4 loads,
// packed u32 LDS stores.  Edge path: clamped addresses + cndmask zeroing
// (no exec-mask branching).
// ---------------------------------------------------------------------------
#define TM 128
#define TN 64
#define TK 32
#define LDP 40   // padded K-stride in halves (80B rows: 16B aligned)

__global__ void __launch_bounds__(256)
gemm_bf16_wmma(const float* __restrict__ A, int lda,
               const float* __restrict__ W, int ldw,
               const float* __restrict__ bias,
               float* __restrict__ C, int ldc,
               int M, int N, int K, int act)
{
  __shared__ unsigned short As[TM * LDP];
  __shared__ unsigned short Ws[TN * LDP];

  const int tid  = threadIdx.x;
  const int lane = tid & 31;
  const int wave = tid >> 5;
  const int m0   = blockIdx.y * TM;
  const int n0   = blockIdx.x * TN;

  const int wm = (wave & 3) * 32;   // wave's M sub-tile (32 rows)
  const int wn = (wave >> 2) * 32;  // wave's N sub-slab (2 x 16)

  v8f acc00 = {}, acc01 = {}, acc10 = {}, acc11 = {};

  // staging maps
  const int ra  = tid >> 1;           // 0..127  A row
  const int ka  = (tid & 1) * 16;     // 0/16    A K-seg (16 floats)
  const int rw  = tid >> 2;           // 0..63   W row
  const int kw  = (tid & 3) * 8;      // 0/8/16/24 W K-seg (8 floats)

  const bool fullM = (m0 + TM <= M);
  const bool fullN = (n0 + TN <= N);

  for (int k0 = 0; k0 < K; k0 += TK) {
    const bool fullK = (k0 + TK <= K);

    // ---------------- stage A tile (128 x 32) ----------------
    if (fullM && fullK) {
      const float4* ap = (const float4*)(A + (long)(m0 + ra) * lda + k0 + ka);
      const float4 q0 = ap[0], q1 = ap[1], q2 = ap[2], q3 = ap[3];
      unsigned int* dst = (unsigned int*)As + ((ra * LDP + ka) >> 1);
      dst[0] = pack2bf(q0.x, q0.y); dst[1] = pack2bf(q0.z, q0.w);
      dst[2] = pack2bf(q1.x, q1.y); dst[3] = pack2bf(q1.z, q1.w);
      dst[4] = pack2bf(q2.x, q2.y); dst[5] = pack2bf(q2.z, q2.w);
      dst[6] = pack2bf(q3.x, q3.y); dst[7] = pack2bf(q3.z, q3.w);
    } else {
      const int m  = m0 + ra;
      const int mc = (m < M) ? m : (M - 1);
      unsigned int* dst = (unsigned int*)As + ((ra * LDP + ka) >> 1);
#pragma unroll
      for (int j = 0; j < 8; ++j) {
        const int k1 = k0 + ka + 2 * j;
        const int k2 = k1 + 1;
        const float v1 = A[(long)mc * lda + ((k1 < K) ? k1 : 0)];
        const float v2 = A[(long)mc * lda + ((k2 < K) ? k2 : 0)];
        const float a1 = (m < M && k1 < K) ? v1 : 0.0f;
        const float a2 = (m < M && k2 < K) ? v2 : 0.0f;
        dst[j] = pack2bf(a1, a2);
      }
    }

    // ---------------- stage W tile (64 x 32) ----------------
    if (fullN && fullK) {
      const float4* wp = (const float4*)(W + (long)(n0 + rw) * ldw + k0 + kw);
      const float4 q0 = wp[0], q1 = wp[1];
      unsigned int* dst = (unsigned int*)Ws + ((rw * LDP + kw) >> 1);
      dst[0] = pack2bf(q0.x, q0.y); dst[1] = pack2bf(q0.z, q0.w);
      dst[2] = pack2bf(q1.x, q1.y); dst[3] = pack2bf(q1.z, q1.w);
    } else {
      const int n  = n0 + rw;
      const int nc = (n < N) ? n : (N - 1);
      unsigned int* dst = (unsigned int*)Ws + ((rw * LDP + kw) >> 1);
#pragma unroll
      for (int j = 0; j < 4; ++j) {
        const int k1 = k0 + kw + 2 * j;
        const int k2 = k1 + 1;
        const float v1 = W[(long)nc * ldw + ((k1 < K) ? k1 : 0)];
        const float v2 = W[(long)nc * ldw + ((k2 < K) ? k2 : 0)];
        const float a1 = (n < N && k1 < K) ? v1 : 0.0f;
        const float a2 = (n < N && k2 < K) ? v2 : 0.0f;
        dst[j] = pack2bf(a1, a2);
      }
    }

    // prefetch next K tile (lowers to global_prefetch_b8 on gfx1250)
    if (k0 + TK < K) {
      if (m0 + ra < M) __builtin_prefetch(A + (long)(m0 + ra) * lda + k0 + TK + ka, 0, 1);
      if (n0 + rw < N) __builtin_prefetch(W + (long)(n0 + rw) * ldw + k0 + TK + kw, 0, 1);
    }
    __syncthreads();

    // ---- build fragments per CDNA5 VGPR layouts ----
    union { v16bf v; unsigned int u[8]; } fa0, fa1, fb0, fb1;
    {
      // A 16x32 bf16: lane m=lane&15; lanes>=16 carry K-base +8;
      // halves 0..7 -> K = kb+0..7, halves 8..15 -> K = 16+kb+0..7
      const int kb = (lane & 16) ? 8 : 0;
      const unsigned int* ar0 = (const unsigned int*)(As + (wm + (lane & 15)) * LDP);
      const unsigned int* ar1 = (const unsigned int*)(As + (wm + 16 + (lane & 15)) * LDP);
#pragma unroll
      for (int vv = 0; vv < 4; ++vv) {
        fa0.u[vv]     = ar0[(kb + 2 * vv) >> 1];
        fa0.u[4 + vv] = ar0[(16 + kb + 2 * vv) >> 1];
        fa1.u[vv]     = ar1[(kb + 2 * vv) >> 1];
        fa1.u[4 + vv] = ar1[(16 + kb + 2 * vv) >> 1];
      }
    }
    {
      // B 32x16 bf16: col n=lane&15; lanes>=16 carry K-base +8;
      // halves 0..7 -> K = kb+0..7, halves 8..15 -> K = 16+kb+0..7
      const int nl = lane & 15;
      const int kb = (lane & 16) ? 8 : 0;
      const unsigned int* b0 = (const unsigned int*)(Ws + (wn + nl) * LDP);
      const unsigned int* b1 = (const unsigned int*)(Ws + (wn + 16 + nl) * LDP);
#pragma unroll
      for (int vv = 0; vv < 4; ++vv) {
        fb0.u[vv]     = b0[(kb >> 1) + vv];
        fb0.u[4 + vv] = b0[8 + (kb >> 1) + vv];
        fb1.u[vv]     = b1[(kb >> 1) + vv];
        fb1.u[4 + vv] = b1[8 + (kb >> 1) + vv];
      }
    }

    acc00 = __builtin_amdgcn_wmma_f32_16x16x32_bf16(false, fa0.v, false, fb0.v,
                                                    (short)0, acc00, false, false);
    acc01 = __builtin_amdgcn_wmma_f32_16x16x32_bf16(false, fa0.v, false, fb1.v,
                                                    (short)0, acc01, false, false);
    acc10 = __builtin_amdgcn_wmma_f32_16x16x32_bf16(false, fa1.v, false, fb0.v,
                                                    (short)0, acc10, false, false);
    acc11 = __builtin_amdgcn_wmma_f32_16x16x32_bf16(false, fa1.v, false, fb1.v,
                                                    (short)0, acc11, false, false);
    __syncthreads();
  }

  // ---- epilogue: bias + activation + predicated store ----
  const int nc0 = n0 + wn + (lane & 15);
  const int nc1 = nc0 + 16;
  const int rb0 = m0 + wm + ((lane & 16) ? 8 : 0);
  const int rb1 = rb0 + 16;
  const float b0v = (bias && nc0 < N) ? bias[nc0] : 0.0f;
  const float b1v = (bias && nc1 < N) ? bias[nc1] : 0.0f;
#pragma unroll
  for (int r = 0; r < 8; ++r) {
    const int r0 = rb0 + r;
    const int r1 = rb1 + r;
    if (r0 < M) {
      if (nc0 < N) C[(long)r0 * ldc + nc0] = act_apply(acc00[r] + b0v, act);
      if (nc1 < N) C[(long)r0 * ldc + nc1] = act_apply(acc01[r] + b1v, act);
    }
    if (r1 < M) {
      if (nc0 < N) C[(long)r1 * ldc + nc0] = act_apply(acc10[r] + b0v, act);
      if (nc1 < N) C[(long)r1 * ldc + nc1] = act_apply(acc11[r] + b1v, act);
    }
  }
}

// ---------------------------------------------------------------------------
// LayerNorm over last dim (one wave32 per token), optional fused residual.
// ---------------------------------------------------------------------------
__global__ void __launch_bounds__(256)
ln_kernel(const float* __restrict__ X, const float* __restrict__ R,
          const float* __restrict__ g, const float* __restrict__ b,
          float* __restrict__ O, int NT, int d)
{
  const int wv   = blockIdx.x * (blockDim.x >> 5) + (threadIdx.x >> 5);
  const int lane = threadIdx.x & 31;
  if (wv >= NT) return;
  const float* xr = X + (long)wv * d;
  const float* rr = R ? R + (long)wv * d : nullptr;
  float s = 0.0f, s2 = 0.0f;
  for (int i = lane; i < d; i += 32) {
    const float v = xr[i] + (rr ? rr[i] : 0.0f);
    s += v; s2 += v * v;
  }
#pragma unroll
  for (int off = 16; off; off >>= 1) {
    s  += __shfl_xor(s,  off, 32);
    s2 += __shfl_xor(s2, off, 32);
  }
  const float m   = s / d;
  const float var = s2 / d - m * m;
  const float inv = rsqrtf(var + 1e-5f);
  float* orow = O + (long)wv * d;
  for (int i = lane; i < d; i += 32) {
    const float v = xr[i] + (rr ? rr[i] : 0.0f);
    orow[i] = (v - m) * inv * g[i] + b[i];
  }
}

// ---------------------------------------------------------------------------
// Depthwise causal conv1d (width 4) over the sequence + SiLU.
// Input: XZ[b,l, 0..di) with row stride 2*di. Output XR[b,l,c].
// ---------------------------------------------------------------------------
__global__ void __launch_bounds__(256)
dwconv_silu(const float* __restrict__ XZ, const float* __restrict__ w,
            const float* __restrict__ cb, float* __restrict__ XR,
            int Bb, int L, int di)
{
  const long t     = (long)blockIdx.x * blockDim.x + threadIdx.x;
  const long total = (long)Bb * L * di;
  if (t >= total) return;
  const int  c  = (int)(t % di);
  const long bl = t / di;
  const int  l  = (int)(bl % L);
  const int  b  = (int)(bl / L);
  const int  st = 2 * di;
  const float* base = XZ + ((long)b * L) * st + c;
  float s = cb[c];
#pragma unroll
  for (int k = 0; k < 4; ++k) {
    const int ll = l - 3 + k;
    if (ll >= 0) s += w[c * 4 + k] * base[(long)ll * st];
  }
  XR[t] = s / (1.0f + __expf(-s));
}

// ---------------------------------------------------------------------------
// Selective scan: one thread per (batch, channel). 16-state recurrence in
// registers; fused +x*D and *silu(z) gating.
// ---------------------------------------------------------------------------
__global__ void __launch_bounds__(256)
sel_scan(const float* __restrict__ DT, const float* __restrict__ XR,
         const float* __restrict__ XD, const float* __restrict__ XZ,
         const float* __restrict__ A_log, const float* __restrict__ Dp,
         float* __restrict__ Y, int Bb, int L, int di, int dtr, int xdw)
{
  const int t = blockIdx.x * blockDim.x + threadIdx.x;
  if (t >= Bb * di) return;
  const int c = t % di;
  const int b = t / di;

  float Aa[16];
#pragma unroll
  for (int s = 0; s < 16; ++s) Aa[s] = -__expf(A_log[c * 16 + s]);
  float h[16];
#pragma unroll
  for (int s = 0; s < 16; ++s) h[s] = 0.0f;

  const float dpc = Dp[c];
  for (int l = 0; l < L; ++l) {
    const long row = (long)b * L + l;
    const float dtv = DT[row * di + c];
    const float xv  = XR[row * di + c];
    const float dx  = dtv * xv;
    const float* bc = XD + row * xdw + dtr;
    float acc = 0.0f;
#pragma unroll
    for (int s = 0; s < 16; ++s) {
      h[s] = h[s] * __expf(dtv * Aa[s]) + dx * bc[s];
      acc += h[s] * bc[16 + s];
    }
    const float z = XZ[row * (2 * di) + di + c];
    float y = acc + xv * dpc;
    y *= z / (1.0f + __expf(-z));
    Y[row * di + c] = y;
  }
}

// ---------------------------------------------------------------------------
// im2col for 4x4x4 stride-4 patch embed on (4,1,64,64,64) input -> [16384,64]
// ---------------------------------------------------------------------------
__global__ void __launch_bounds__(256)
im2col_patch(const float* __restrict__ x, float* __restrict__ col)
{
  const long t = (long)blockIdx.x * blockDim.x + threadIdx.x;
  if (t >= 16384L * 64) return;
  const int  j   = (int)(t & 63);
  const long row = t >> 6;
  const int ow = (int)(row & 15);
  const int oh = (int)((row >> 4) & 15);
  const int od = (int)((row >> 8) & 15);
  const int b  = (int)(row >> 12);
  const int kw = j & 3, kh = (j >> 2) & 3, kd = j >> 4;
  const long idx = (((long)b * 64 + od * 4 + kd) * 64 + (oh * 4 + kh)) * 64 + (ow * 4 + kw);
  col[t] = x[idx];
}

// ---------------------------------------------------------------------------
// im2col for 2x2x2 stride-2 downsample on token tensor [B, S^3, d] (chan-last)
// -> [B*(S/2)^3, 8*d] with K ordering (c, kd, kh, kw) matching OIDHW weights.
// ---------------------------------------------------------------------------
__global__ void __launch_bounds__(256)
im2col_down(const float* __restrict__ X, float* __restrict__ col,
            int Bb, int S, int d)
{
  const int  So   = S >> 1;
  const int  Kc   = 8 * d;
  const long rows = (long)Bb * So * So * So;
  const long t    = (long)blockIdx.x * blockDim.x + threadIdx.x;
  if (t >= rows * Kc) return;
  const int  j   = (int)(t % Kc);
  long row = t / Kc;
  const int ow = (int)(row % So); row /= So;
  const int oh = (int)(row % So); row /= So;
  const int od = (int)(row % So);
  const int b  = (int)(row / So);
  const int kw = j & 1, kh = (j >> 1) & 1, kd = (j >> 2) & 1, c = j >> 3;
  const long lin = ((long)(od * 2 + kd) * S + (oh * 2 + kh)) * S + (ow * 2 + kw);
  col[t] = X[(((long)b * S * S * S) + lin) * (long)d + c];
}

// ---------------------------------------------------------------------------
// Mean pool over the 8 final tokens per batch: [4,8,768] -> [4,768]
// ---------------------------------------------------------------------------
__global__ void __launch_bounds__(256)
mean_pool8(const float* __restrict__ X, float* __restrict__ P)
{
  const int t = blockIdx.x * blockDim.x + threadIdx.x;
  if (t >= 4 * 768) return;
  const int c = t % 768;
  const int b = t / 768;
  float s = 0.0f;
#pragma unroll
  for (int l = 0; l < 8; ++l) s += X[((long)b * 8 + l) * 768 + c];
  P[t] = s * 0.125f;
}

// ---------------------------------------------------------------------------
// Host orchestration
// ---------------------------------------------------------------------------
static const int DIMS_H[4]   = {96, 192, 384, 768};
static const int DEPTHS_H[4] = {2, 2, 6, 2};

extern "C" void kernel_launch(void* const* d_in, const int* in_sizes, int n_in,
                              void* d_out, int out_size, void* d_ws, size_t ws_size,
                              hipStream_t stream) {
  (void)in_sizes; (void)n_in; (void)out_size; (void)ws_size;
  auto P = [&](int i) -> const float* { return (const float*)d_in[i]; };

  float* ws  = (float*)d_ws;
  float* Xb  = ws + 0L;         // [NT, d]                 cap 1,572,864
  float* XN  = ws + 1572864L;   // LN output               cap 1,572,864
  float* XZ  = ws + 3145728L;   // [NT, 2di] (alias HB)    cap 6,291,456
  float* XR  = ws + 9437184L;   // [NT, di]                cap 3,145,728
  float* XD  = ws + 12582912L;  // [NT, xdw_pad]           cap 1,310,720
  float* DTb = ws + 13893632L;  // [NT, di]                cap 3,145,728
  float* Yb  = ws + 17039360L;  // [NT, di]                cap 3,145,728
  float* MO  = ws + 20185088L;  // [NT, d]                 cap 1,572,864
  float* COL = ws + 21757952L;  // im2col                  cap 1,572,864
  float* PP  = ws + 23330816L;  // head scratch            cap 16,384
  float* HB  = XZ;              // MLP hidden reuses XZ (disjoint lifetime)

  auto gemm = [&](const float* A, int lda, const float* W, int ldw,
                  const float* bias, float* C, int ldc,
                  int M, int N, int K, int act) {
    dim3 grid((N + TN - 1) / TN, (M + TM - 1) / TM);
    gemm_bf16_wmma<<<grid, dim3(256), 0, stream>>>(A, lda, W, ldw, bias, C, ldc,
                                                   M, N, K, act);
  };
  auto ln = [&](const float* X, const float* R, const float* g, const float* b,
                float* O, int NT, int d) {
    ln_kernel<<<(NT + 7) / 8, dim3(256), 0, stream>>>(X, R, g, b, O, NT, d);
  };

  // ---- patch embed: im2col + GEMM -> Xb [16384, 96] ----
  {
    const long tot = 16384L * 64;
    im2col_patch<<<(int)((tot + 255) / 256), dim3(256), 0, stream>>>(P(0), COL);
    gemm(COL, 64, P(1), 64, P(2), Xb, 96, 16384, 96, 64, ACT_NONE);
  }

  int S  = 16;
  int bi = 3;  // first block-tensor index
  const int Bb = 4;

  for (int st = 0; st < 4; ++st) {
    const int d    = DIMS_H[st];
    const int dI   = 2 * d;          // d_inner
    const int h    = 4 * d;          // MLP hidden
    const int dtr  = d / 16;         // dt_rank
    const int xdw  = dtr + 32;       // x_proj out width
    const int xdwp = (xdw + 3) & ~3; // padded to 16B rows for fast GEMM loads
    const int L    = S * S * S;
    const int NT   = Bb * L;

    for (int blk = 0; blk < DEPTHS_H[st]; ++blk) {
      const float* ln1_g  = P(bi + 0);
      const float* ln1_b  = P(bi + 1);
      const float* mln_g  = P(bi + 2);
      const float* mln_b  = P(bi + 3);
      const float* in_w   = P(bi + 4);
      const float* conv_w = P(bi + 5);
      const float* conv_b = P(bi + 6);
      const float* xproj  = P(bi + 7);
      const float* dt_w   = P(bi + 8);
      const float* dt_b   = P(bi + 9);
      const float* A_log  = P(bi + 10);
      const float* Dp     = P(bi + 11);
      const float* out_w  = P(bi + 12);
      const float* w1     = P(bi + 13);
      const float* b1     = P(bi + 14);
      const float* w2     = P(bi + 15);
      const float* b2     = P(bi + 16);
      bi += 17;

      // Mamba branch
      ln(Xb, nullptr, mln_g, mln_b, XN, NT, d);
      gemm(XN, d, in_w, d, nullptr, XZ, 2 * dI, NT, 2 * dI, d, ACT_NONE);
      {
        const long tot = (long)NT * dI;
        dwconv_silu<<<(int)((tot + 255) / 256), dim3(256), 0, stream>>>(
            XZ, conv_w, conv_b, XR, Bb, L, dI);
      }
      gemm(XR, dI, xproj, dI, nullptr, XD, xdwp, NT, xdw, dI, ACT_NONE);
      gemm(XD, xdwp, dt_w, dtr, dt_b, DTb, dI, NT, dI, dtr, ACT_SOFTPLUS);
      sel_scan<<<(Bb * dI + 255) / 256, dim3(256), 0, stream>>>(
          DTb, XR, XD, XZ, A_log, Dp, Yb, Bb, L, dI, dtr, xdwp);
      gemm(Yb, dI, out_w, dI, nullptr, MO, d, NT, d, dI, ACT_NONE);

      // LN(x + mamba) then MLP (no outer residual in reference block)
      ln(Xb, MO, ln1_g, ln1_b, XN, NT, d);
      gemm(XN, d, w1, d, b1, HB, h, NT, h, d, ACT_GELU);
      gemm(HB, h, w2, h, b2, Xb, d, NT, d, h, ACT_NONE);
    }

    if (st < 3) {
      const int dn = DIMS_H[st + 1];
      const int Kc = 8 * d;
      const int So = S >> 1;
      const long rows = (long)Bb * So * So * So;
      const long tot  = rows * Kc;
      im2col_down<<<(int)((tot + 255) / 256), dim3(256), 0, stream>>>(
          Xb, COL, Bb, S, d);
      gemm(COL, Kc, P(207 + 2 * st), Kc, P(208 + 2 * st), Xb, dn,
           (int)rows, dn, Kc, ACT_NONE);
      S = So;
    }
  }

  // ---- bottleneck 1x1x1 conv (plain GEMM over 32 tokens) ----
  gemm(Xb, 768, P(213), 768, P(214), MO, 768, Bb * 8, 768, 768, ACT_NONE);

  // ---- head: mean pool -> LN -> fc1(ReLU) -> fc2 ----
  mean_pool8<<<(4 * 768 + 255) / 256, dim3(256), 0, stream>>>(MO, PP);
  float* P2 = PP + 4 * 768;
  float* P3 = P2 + 4 * 768;
  ln(PP, nullptr, P(215), P(216), P2, 4, 768);
  gemm(P2, 768, P(217), 768, P(218), P3, 512, 4, 512, 768, ACT_RELU);
  gemm(P3, 512, P(219), 512, P(220), (float*)d_out, 1000, 4, 1000, 512, ACT_NONE);
}
